// BahdanauAttention_21268678049849
// MI455X (gfx1250) — matrix operand order repaired and checked by reading, not compiled, not run on hardware
//
#include <hip/hip_runtime.h>
#include <hip/hip_bf16.h>

typedef __bf16 bf16;
typedef __attribute__((ext_vector_type(16))) __bf16 v16bf;
typedef __attribute__((ext_vector_type(8)))  __bf16 v8bf;
typedef __attribute__((ext_vector_type(4)))  __bf16 v4bf;
typedef __attribute__((ext_vector_type(8)))  float  v8f;
typedef __attribute__((ext_vector_type(4)))  int    v4i;

#define SEQ   2048
#define BATCH 32
#define KDIM  1024   // 2*ENC_HS (GEMM K)
#define DDIM  1024   // DEC_HS   (GEMM N / energy dim)
#define ROWS  (SEQ * BATCH)

#define BR    64     // enc rows per block
#define TPB   256    // 8 waves
#define LDA   1032   // padded LDS stride in bf16 elements (1024 + 8)
#define NCH   (DDIM / 32)          // 32 N-chunks of 32 columns
#define WCPY  (32 * (KDIM / 8))    // v8bf chunks per W stage (4096)

#define AS1 __attribute__((address_space(1)))
#define AS3 __attribute__((address_space(3)))

#if defined(__HIP_DEVICE_COMPILE__) && defined(__has_builtin)
#if __has_builtin(__builtin_amdgcn_global_load_async_to_lds_b128) && \
    __has_builtin(__builtin_amdgcn_s_wait_asynccnt)
#define USE_ASYNC_LDS 1
#endif
#if __has_builtin(__builtin_amdgcn_s_barrier_signal) && \
    __has_builtin(__builtin_amdgcn_s_barrier_wait)
#define USE_SPLIT_BARRIER 1
#endif
#endif

#if USE_ASYNC_LDS
__device__ __forceinline__ void async_b128(const void* g, void* l) {
    __builtin_amdgcn_global_load_async_to_lds_b128(
        (AS1 v4i*)(AS1 void*)(g),
        (AS3 v4i*)(AS3 void*)(l), 0, 0);
}
#endif

__device__ __forceinline__ void wg_barrier_signal() {
#if USE_SPLIT_BARRIER
    __builtin_amdgcn_s_barrier_signal(-1);
#endif
}
__device__ __forceinline__ void wg_barrier_wait() {
#if USE_SPLIT_BARRIER
    __builtin_amdgcn_s_barrier_wait(-1);
#else
    __syncthreads();
#endif
}

__device__ __forceinline__ float fast_tanhf(float x) {
#if defined(__has_builtin) && __has_builtin(__builtin_amdgcn_tanhf)
    return __builtin_amdgcn_tanhf(x);
#else
    // branch-free: tanh(x) = 1 - 2/(exp(2x)+1); saturates correctly at +-inf
    const float e = __expf(2.0f * x);
    return 1.0f - 2.0f / (e + 1.0f);
#endif
}

// ---------------------------------------------------------------------------
// Kernel 1: transpose-convert W_e (rows 1024..2047 of attn_w) into WeT[n][k] bf16
// ---------------------------------------------------------------------------
__global__ void we_transpose(const float* __restrict__ W, bf16* __restrict__ WeT) {
    __shared__ float tile[32][33];
    const int kb = blockIdx.x * 32;
    const int nb = blockIdx.y * 32;
    for (int j = threadIdx.y; j < 32; j += 8)
        tile[j][threadIdx.x] = W[(size_t)(KDIM + kb + j) * DDIM + nb + threadIdx.x];
    __syncthreads();
    for (int j = threadIdx.y; j < 32; j += 8)
        WeT[(size_t)(nb + j) * KDIM + kb + threadIdx.x] = (bf16)tile[threadIdx.x][j];
}

// ---------------------------------------------------------------------------
// Kernel 2: preT[n][b] = sum_k hidden[b][k] * attn_w[k][n] + attn_b[n]  (fp32,
// stored n-major so the WMMA epilogue reads consecutive b with float4)
// ---------------------------------------------------------------------------
__global__ void pre_kernel(const float* __restrict__ hidden,
                           const float* __restrict__ W,
                           const float* __restrict__ bias,
                           float* __restrict__ preT) {
    __shared__ float h[DDIM];
    const int b = blockIdx.y;
    const int n = blockIdx.x * 128 + threadIdx.x;
    for (int i = threadIdx.x; i < DDIM; i += 128) h[i] = hidden[(size_t)b * DDIM + i];
    __syncthreads();
    float acc = bias[n];
    #pragma unroll 8
    for (int k = 0; k < DDIM; ++k)
        acc = fmaf(h[k], W[(size_t)k * DDIM + n], acc);
    preT[(size_t)n * BATCH + b] = acc;
}

// ---------------------------------------------------------------------------
// Kernel 3: fused  scores[b][s] = v . tanh( enc_row @ W_e + pre[b] )
// ---------------------------------------------------------------------------
__launch_bounds__(TPB, 1)
__global__ void attn_scores(const float* __restrict__ enc,
                            const bf16*  __restrict__ WeT,
                            const float* __restrict__ preT,
                            const float* __restrict__ vvec,
                            float* __restrict__ scores) {
    extern __shared__ char smem[];
    bf16*  As  = (bf16*)smem;                 // [BR][LDA]
    bf16*  Ws0 = As + BR * LDA;               // [32][LDA]  double buffer 0
    bf16*  Ws1 = Ws0 + 32 * LDA;              // [32][LDA]  double buffer 1
    float* red = (float*)(Ws1 + 32 * LDA);    // [8][16]

    const int tid  = threadIdx.x;
    const int lane = tid & 31;
    const int wave = tid >> 5;
    const int frag = wave & 3;    // which 16-row A fragment (0..3)
    const int nsub = wave >> 2;   // which 16-col half of the 32-col W chunk (0..1)
    const int nl   = lane & 15;
    const int hs   = lane >> 4;   // 0: lanes 0-15, 1: lanes 16-31
    const size_t rowBase = (size_t)blockIdx.x * BR;

#if USE_ASYNC_LDS
    // kick off chunk 0 staging before touching A (ASYNCcnt tracks completion)
    #pragma unroll
    for (int c = tid; c < WCPY; c += TPB) {
        const int n = c >> 7, kc = (c & 127) << 3;
        async_b128(WeT + (size_t)n * KDIM + kc, Ws0 + n * LDA + kc);
    }
#endif

    // ---- stage A tile: 64 rows x 1024 fp32 -> bf16 in LDS ----
    for (int c = tid; c < BR * (KDIM / 4); c += TPB) {
        const int r  = c >> 8;
        const int kc = (c & 255) << 2;
        const float4 f = *(const float4*)(enc + (rowBase + r) * KDIM + kc);
        v4bf d4 = {(bf16)f.x, (bf16)f.y, (bf16)f.z, (bf16)f.w};
        *(v4bf*)(As + r * LDA + kc) = d4;
    }

    const bf16* Arow = As + (frag * 16 + nl) * LDA;
    const int   b0   = (frag & 1) * 16 + hs * 8;   // rowBase%32==0 always
    float partial[8] = {0.f, 0.f, 0.f, 0.f, 0.f, 0.f, 0.f, 0.f};

    for (int nt = 0; nt < NCH; ++nt) {
        const bf16* WsCur;
#if USE_ASYNC_LDS
        WsCur = (nt & 1) ? Ws1 : Ws0;
        if (nt > 0) wg_barrier_wait();   // everyone done reading buffer 'nxt'
        if (nt + 1 < NCH) {
            bf16* nxt = (nt & 1) ? Ws0 : Ws1;
            const bf16* src = WeT + (size_t)(nt + 1) * 32 * KDIM;
            #pragma unroll
            for (int c = tid; c < WCPY; c += TPB) {
                const int n = c >> 7, kc = (c & 127) << 3;
                async_b128(src + (size_t)n * KDIM + kc, nxt + n * LDA + kc);
            }
            // async loads complete in order: <=16 outstanding means the 16
            // loads of the current chunk have landed in LDS
            __builtin_amdgcn_s_wait_asynccnt(16);
        } else {
            __builtin_amdgcn_s_wait_asynccnt(0);
        }
        __syncthreads();   // all waves' async data for 'cur' has landed
#else
        WsCur = Ws0;
        __syncthreads();   // previous chunk fully consumed
        const bf16* src = WeT + (size_t)nt * 32 * KDIM;
        for (int c = tid; c < WCPY; c += TPB) {
            const int n = c >> 7, kc = (c & 127) << 3;
            *(v8bf*)(Ws0 + n * LDA + kc) = *(const v8bf*)(src + (size_t)n * KDIM + kc);
        }
        __syncthreads();
#endif

        v8f acc = {0.f, 0.f, 0.f, 0.f, 0.f, 0.f, 0.f, 0.f};
        const bf16* Bcol = WsCur + (nsub * 16 + nl) * LDA;

        // ---- software-pipelined K loop: stage kk+1 loads behind WMMA kk ----
        v8bf aL[2], aH[2], bL[2], bH[2];
        {
            const int ka0 = hs * 8, kb0 = hs * 8;
            aL[0] = *(const v8bf*)(Arow + ka0);
            aH[0] = *(const v8bf*)(Arow + ka0 + 16);
            bL[0] = *(const v8bf*)(Bcol + kb0);
            bH[0] = *(const v8bf*)(Bcol + kb0 + 16);
        }
        #pragma unroll
        for (int kk = 0; kk < KDIM / 32; ++kk) {
            const int cur = kk & 1, nx = cur ^ 1;
            if (kk + 1 < KDIM / 32) {
                const int ka = (kk + 1) * 32 + hs * 8;
                const int kb = (kk + 1) * 32 + hs * 8;
                aL[nx] = *(const v8bf*)(Arow + ka);
                aH[nx] = *(const v8bf*)(Arow + ka + 16);
                bL[nx] = *(const v8bf*)(Bcol + kb);
                bH[nx] = *(const v8bf*)(Bcol + kb + 16);
            }
            v16bf af  = __builtin_shufflevector(aL[cur], aH[cur],
                            0,1,2,3,4,5,6,7,8,9,10,11,12,13,14,15);
            v16bf bfv = __builtin_shufflevector(bL[cur], bH[cur],
                            0,1,2,3,4,5,6,7,8,9,10,11,12,13,14,15);
            acc = __builtin_amdgcn_wmma_f32_16x16x32_bf16(
                      false, af, false, bfv, (short)0, acc, false, false);
        }

#if USE_ASYNC_LDS
        wg_barrier_signal();   // done reading 'cur'; epilogue overlaps others
#endif

        // ---- fused epilogue: tanh + v-dot, accumulate per row ----
        const int   ng = nt * 32 + nsub * 16 + nl;
        const float vv = vvec[ng];
        const float4 p0 = *(const float4*)(preT + (size_t)ng * BATCH + b0);
        const float4 p1 = *(const float4*)(preT + (size_t)ng * BATCH + b0 + 4);
        partial[0] += vv * fast_tanhf(acc[0] + p0.x);
        partial[1] += vv * fast_tanhf(acc[1] + p0.y);
        partial[2] += vv * fast_tanhf(acc[2] + p0.z);
        partial[3] += vv * fast_tanhf(acc[3] + p0.w);
        partial[4] += vv * fast_tanhf(acc[4] + p1.x);
        partial[5] += vv * fast_tanhf(acc[5] + p1.y);
        partial[6] += vv * fast_tanhf(acc[6] + p1.z);
        partial[7] += vv * fast_tanhf(acc[7] + p1.w);

#if !USE_ASYNC_LDS
        __syncthreads();   // chunk consumed; next iteration may overwrite
#endif
    }

#if USE_ASYNC_LDS
    wg_barrier_wait();     // drain the last signaled round (keeps pairing)
#endif

    // ---- reduce over the 16 lanes holding one row's columns ----
    #pragma unroll
    for (int r = 0; r < 8; ++r) {
        float s = partial[r];
        s += __shfl_xor(s, 1, 32);
        s += __shfl_xor(s, 2, 32);
        s += __shfl_xor(s, 4, 32);
        s += __shfl_xor(s, 8, 32);
        partial[r] = s;
    }
    if (nl == 0) {
        #pragma unroll
        for (int r = 0; r < 8; ++r)
            red[wave * 16 + hs * 8 + r] = partial[r];
    }
    __syncthreads();
    // waves f and f+4 covered the two 16-col halves of the same rows: sum them
    if (tid < BR) {
        const int f = tid >> 4, m = tid & 15;
        const float s = red[f * 16 + m] + red[(f + 4) * 16 + m];
        const size_t rowG = rowBase + f * 16 + m;
        const int b    = (int)(rowG & (BATCH - 1));
        const int sIdx = (int)(rowG >> 5);
        scores[(size_t)b * SEQ + sIdx] = s;
    }
}

// ---------------------------------------------------------------------------
// Kernel 4: softmax over S per batch row
// ---------------------------------------------------------------------------
__global__ void softmax_rows(const float* __restrict__ scores, float* __restrict__ out) {
    __shared__ float red[256];
    const int b = blockIdx.x, tid = threadIdx.x;
    const float* row = scores + (size_t)b * SEQ;

    float m = -3.4e38f;
    for (int s = tid; s < SEQ; s += 256) m = fmaxf(m, row[s]);
    red[tid] = m; __syncthreads();
    for (int off = 128; off > 0; off >>= 1) {
        if (tid < off) red[tid] = fmaxf(red[tid], red[tid + off]);
        __syncthreads();
    }
    m = red[0];
    __syncthreads();

    float sum = 0.f;
    for (int s = tid; s < SEQ; s += 256) sum += __expf(row[s] - m);
    red[tid] = sum; __syncthreads();
    for (int off = 128; off > 0; off >>= 1) {
        if (tid < off) red[tid] += red[tid + off];
        __syncthreads();
    }
    const float inv = 1.0f / red[0];
    for (int s = tid; s < SEQ; s += 256)
        out[(size_t)b * SEQ + s] = __expf(row[s] - m) * inv;
}

// ---------------------------------------------------------------------------
extern "C" void kernel_launch(void* const* d_in, const int* in_sizes, int n_in,
                              void* d_out, int out_size, void* d_ws, size_t ws_size,
                              hipStream_t stream) {
    const float* hidden = (const float*)d_in[0];   // [32,1024]
    const float* enc    = (const float*)d_in[1];   // [2048,32,1024] -> rows (s*32+b)
    const float* attn_w = (const float*)d_in[2];   // [2048,1024]
    const float* attn_b = (const float*)d_in[3];   // [1024]
    const float* v      = (const float*)d_in[4];   // [1024]

    char* ws = (char*)d_ws;
    bf16*  WeT    = (bf16*)ws;                                   // 2 MB
    float* preT   = (float*)(ws + (size_t)2 * 1024 * 1024);      // 128 KB
    float* scores = (float*)(ws + (size_t)2 * 1024 * 1024 + 128 * 1024); // 256 KB

    we_transpose<<<dim3(KDIM / 32, DDIM / 32), dim3(32, 8), 0, stream>>>(attn_w, WeT);
    pre_kernel<<<dim3(DDIM / 128, BATCH), 128, 0, stream>>>(hidden, attn_w, attn_b, preT);

    // LDS: A tile + both W double-buffers + reduction scratch (host cannot see
    // the device-side __has_builtin, so always size for the async path)
    const size_t lds = (size_t)(BR + 64) * LDA * sizeof(bf16) + 8 * 16 * sizeof(float);
    attn_scores<<<ROWS / BR, TPB, lds, stream>>>(enc, WeT, preT, v, scores);

    softmax_rows<<<BATCH, 256, 0, stream>>>(scores, (float*)d_out);
}
